// TransformerConv_62758062129889
// MI455X (gfx1250) — compile-verified, matrix operand order checked by reading, not run on hardware
//
#include <hip/hip_runtime.h>
#include <math.h>

#define NN   20000
#define BB   2
#define FIN  256
#define EE   320000
#define EDIM 64
#define HH   8
#define CC   32
#define HC   256           // H*C
#define MROWS (NN*BB)      // 40000

typedef __attribute__((ext_vector_type(16))) _Float16 v16h;
typedef __attribute__((ext_vector_type(8)))  _Float16 v8h;
typedef __attribute__((ext_vector_type(2)))  _Float16 v2h;
typedef __attribute__((ext_vector_type(8)))  float    v8f;

union AV { v16h v; struct { v8h lo, hi; } p; };

// ---- monotone float<->uint encoding for atomicMax-based segment max ----
__device__ __forceinline__ unsigned enc_f32(float f){
  unsigned u = __float_as_uint(f);
  return (u & 0x80000000u) ? ~u : (u | 0x80000000u);
}
__device__ __forceinline__ float dec_f32(unsigned u){
  return (u & 0x80000000u) ? __uint_as_float(u & 0x7FFFFFFFu) : __uint_as_float(~u);
}

// ---------------- init softmax state ----------------
__global__ void init_kernel(unsigned* __restrict__ menc, float* __restrict__ z){
  int i = blockIdx.x*blockDim.x + threadIdx.x;
  if (i < NN*BB*HH){ menc[i] = 0u; z[i] = 0.f; }   // enc==0 is below every real logit
}

// ---------------- one-off: x (f32) -> xh (f16), 2 elements/thread ----------------
__global__ void cvtx_kernel(const float* __restrict__ x, _Float16* __restrict__ xh){
  int i = blockIdx.x*blockDim.x + threadIdx.x;     // over (M*HC)/2, grid exact
  float2 xv = ((const float2*)x)[i];
  v2h o; o[0] = (_Float16)xv.x; o[1] = (_Float16)xv.y;
  ((v2h*)xh)[i] = o;
}

// ---------------- one-off: transpose+convert the 4 weights into f16 [4*256,256] ------
__global__ void cvtw_kernel(const float* __restrict__ Wq, const float* __restrict__ Wk,
                            const float* __restrict__ Wv, const float* __restrict__ Wr,
                            _Float16* __restrict__ WhT){
  const float* W = (blockIdx.y==0)?Wq:(blockIdx.y==1)?Wk:(blockIdx.y==2)?Wv:Wr;
  int r = blockIdx.x;          // K row 0..255
  int c = threadIdx.x;         // col   0..255
  WhT[blockIdx.y*65536 + c*256 + r] = (_Float16)W[r*256 + c];
}

// ---------------- one-off: pack the 4 bias vectors into bcat[1024] ----------------
__global__ void bias_kernel(const float* __restrict__ bq, const float* __restrict__ bk,
                            const float* __restrict__ bv, const float* __restrict__ br,
                            float* __restrict__ bcat){
  int i = blockIdx.x*blockDim.x + threadIdx.x;     // 0..1023
  int mat = i >> 8, n = i & 255;
  const float* b = (mat==0)?bq:(mat==1)?bk:(mat==2)?bv:br;
  bcat[i] = b[n];
}

// ---------------- WeSum[d,h] = sum_c We[d, h*32+c]  (e only used via sum over C) ----
__global__ void wesum_kernel(const float* __restrict__ We, float* __restrict__ WeSum){
  int t = threadIdx.x;                 // 0..511
  int d = t >> 3, h = t & 7;
  float s = 0.f;
  #pragma unroll
  for (int c = 0; c < CC; ++c) s += We[d*HC + h*CC + c];
  WeSum[d*HH + h] = s;
}

// ---------------- esum[E,H] = edata @ WeSum ----------------
__global__ void esum_kernel(const float* __restrict__ edata, const float* __restrict__ WeSum,
                            float* __restrict__ esum){
  __shared__ float ws[EDIM*HH];
  for (int i = threadIdx.x; i < EDIM*HH; i += blockDim.x) ws[i] = WeSum[i];
  __syncthreads();
  int idx = blockIdx.x*blockDim.x + threadIdx.x;   // = e*8 + h, grid sized exactly
  int ed  = idx >> 3, h = idx & 7;
  const float* row = edata + (size_t)ed*EDIM;
  float s = 0.f;
  #pragma unroll
  for (int d = 0; d < EDIM; ++d) s += row[d]*ws[d*HH + h];
  esum[idx] = s;
}

// ---------------- fused WMMA GEMM: {q,k,v,r}[40000,256] = xh @ WhT^T + bcat ----------
// One wave computes a 16x64 strip: 4 independent accumulators, A fragment reused 4x,
// 32x v_wmma_f32_16x16x32_f16 per wave with no single dependency chain.
__global__ void gemm_wmma(const _Float16* __restrict__ xh, const _Float16* __restrict__ WhT,
                          const float* __restrict__ bcat,
                          float* __restrict__ oq, float* __restrict__ ok_,
                          float* __restrict__ ov, float* __restrict__ orr){
  int wave  = threadIdx.x >> 5;
  int lane  = threadIdx.x & 31;
  int mtile = blockIdx.x*4 + wave;     // 0..2499 (exact)
  int strip = blockIdx.y;              // 0..15 over fused N=1024; 64 cols per strip
  int lh    = lane >> 4;               // K-half selector per ISA A/B layout
  const _Float16* a0 = xh  + (size_t)(mtile*16 + (lane & 15))*FIN;
  const _Float16* b0 = WhT + (size_t)(strip*64 + (lane & 15))*FIN;
  v8f acc[4] = {v8f{}, v8f{}, v8f{}, v8f{}};
  #pragma unroll
  for (int ko = 0; ko < FIN; ko += 32){
    int b1 = ko + lh*8;                // contiguous 8 K-halves (VGPRs 0..3)
    int b2 = b1 + 16;                  // contiguous 8 K-halves (VGPRs 4..7)
    AV a;
    a.p.lo = *(const v8h*)(a0 + b1);
    a.p.hi = *(const v8h*)(a0 + b2);
    #pragma unroll
    for (int s = 0; s < 4; ++s){
      const _Float16* bp = b0 + (size_t)s*16*FIN;
      AV b;
      b.p.lo = *(const v8h*)(bp + b1);
      b.p.hi = *(const v8h*)(bp + b2);
      acc[s] = __builtin_amdgcn_wmma_f32_16x16x32_f16(false, a.v, false, b.v,
                                                      (short)0, acc[s], false, false);
    }
  }
  // C/D layout: VGPR i -> row (i + 8*lh), col (lane&15)
  int mat = strip >> 2;                               // uniform per block
  float* O = (mat==0)?oq:(mat==1)?ok_:(mat==2)?ov:orr;
  int rowb = mtile*16 + lh*8;
  #pragma unroll
  for (int s = 0; s < 4; ++s){
    int gcol = strip*64 + s*16 + (lane & 15);         // fused column (bias index)
    int col  = (strip & 3)*64 + s*16 + (lane & 15);   // column within matrix
    float bs = bcat[gcol];
    float* op = O + (size_t)rowb*HC + col;
    #pragma unroll
    for (int i = 0; i < 8; ++i) op[(size_t)i*HC] = acc[s][i] + bs;
  }
}

// ---------------- logits + segment max (wave per edge; lane = channel c) ----------------
__global__ void logits_kernel(const float* __restrict__ q, const float* __restrict__ k,
                              const float* __restrict__ esum,
                              const int* __restrict__ esrc, const int* __restrict__ edst,
                              float* __restrict__ logits, unsigned* __restrict__ menc){
  int wave = threadIdx.x >> 5;
  int lane = threadIdx.x & 31;
  int ed   = blockIdx.x*(blockDim.x >> 5) + wave;   // grid exact: E/8 blocks of 8 waves
  int src = esrc[ed], dst = edst[ed];
  const float scale = 0.17677669529663687f;         // 1/sqrt(32)
  #pragma unroll
  for (int b = 0; b < BB; ++b){
    const float* kp = k + (size_t)(src*BB + b)*HC;
    const float* qp = q + (size_t)(dst*BB + b)*HC;
    #pragma unroll
    for (int h = 0; h < HH; ++h){
      float p = kp[h*CC + lane] * qp[h*CC + lane];
      #pragma unroll
      for (int off = 16; off; off >>= 1) p += __shfl_xor(p, off, 32);
      if (lane == 0){
        float lg = p*scale + esum[ed*HH + h];
        logits[((size_t)ed*BB + b)*HH + h] = lg;
        atomicMax(&menc[(dst*BB + b)*HH + h], enc_f32(lg));
      }
    }
  }
}

// ---------------- exp(logit - max), segment sum ----------------
__global__ void exp_kernel(const int* __restrict__ edst, const unsigned* __restrict__ menc,
                           float* __restrict__ logits, float* __restrict__ z){
  int idx = blockIdx.x*blockDim.x + threadIdx.x;    // E*B*H, grid exact
  int ed  = idx / (BB*HH);
  int bh  = idx % (BB*HH);
  int dst = edst[ed];
  float m  = dec_f32(menc[dst*BB*HH + bh]);
  float ex = __expf(logits[idx] - m);
  logits[idx] = ex;                                 // overwrite in place
  atomicAdd(&z[dst*BB*HH + bh], ex);
}

// ---------------- weighted scatter: out[dst] += alpha * v[src]  (wave per edge) ----------
__global__ void agg_kernel(const float* __restrict__ v, const float* __restrict__ logits,
                           const float* __restrict__ z,
                           const int* __restrict__ esrc, const int* __restrict__ edst,
                           float* __restrict__ out){
  int wave = threadIdx.x >> 5;
  int lane = threadIdx.x & 31;
  int ed   = blockIdx.x*(blockDim.x >> 5) + wave;
  int src = esrc[ed], dst = edst[ed];
  #pragma unroll
  for (int b = 0; b < BB; ++b){
    const float* vp = v   + (size_t)(src*BB + b)*HC;
    float*       op = out + (size_t)(dst*BB + b)*HC;
    #pragma unroll
    for (int h = 0; h < HH; ++h){
      float ex = logits[((size_t)ed*BB + b)*HH + h];
      float zz = z[(dst*BB + b)*HH + h];
      float alpha = ex / zz;
      atomicAdd(&op[h*CC + lane], vp[h*CC + lane]*alpha);
    }
  }
}

extern "C" void kernel_launch(void* const* d_in, const int* in_sizes, int n_in,
                              void* d_out, int out_size, void* d_ws, size_t ws_size,
                              hipStream_t stream){
  (void)in_sizes; (void)n_in; (void)out_size; (void)ws_size;
  const float* x     = (const float*)d_in[0];
  const float* edata = (const float*)d_in[1];
  const int*   esrc  = (const int*)  d_in[2];
  const int*   edst  = (const int*)  d_in[3];
  const float* Wq = (const float*)d_in[4];
  const float* bq = (const float*)d_in[5];
  const float* Wk = (const float*)d_in[6];
  const float* bk = (const float*)d_in[7];
  const float* Wv = (const float*)d_in[8];
  const float* bv = (const float*)d_in[9];
  const float* We = (const float*)d_in[10];
  const float* Wr = (const float*)d_in[11];
  const float* br = (const float*)d_in[12];
  float* out = (float*)d_out;

  // workspace layout (float units)
  float*    qb    = (float*)d_ws;
  float*    kb    = qb + (size_t)MROWS*HC;            // 10.24M floats each
  float*    vb    = kb + (size_t)MROWS*HC;
  float*    lg    = vb + (size_t)MROWS*HC;            // E*B*H = 5.12M
  float*    es    = lg + (size_t)EE*BB*HH;            // E*H   = 2.56M
  float*    zb    = es + (size_t)EE*HH;               // N*B*H = 0.32M
  unsigned* me    = (unsigned*)(zb + (size_t)NN*BB*HH);
  float*    bcat  = (float*)(me + (size_t)NN*BB*HH);  // 1024
  float*    WeSum = bcat + 1024;                      // 512
  _Float16* xh    = (_Float16*)(WeSum + 512);         // M*HC halves = 10.24M halves
  _Float16* WhT   = xh + (size_t)MROWS*HC;            // 4*65536 halves

  init_kernel<<<dim3((NN*BB*HH + 255)/256), 256, 0, stream>>>(me, zb);
  cvtx_kernel<<<dim3((size_t)MROWS*HC/2/256), 256, 0, stream>>>(x, xh);
  cvtw_kernel<<<dim3(256, 4), 256, 0, stream>>>(Wq, Wk, Wv, Wr, WhT);
  bias_kernel<<<dim3(4), 256, 0, stream>>>(bq, bk, bv, br, bcat);
  wesum_kernel<<<1, 512, 0, stream>>>(We, WeSum);

  // fused q/k/v/r projection: 2500 M-tiles x 16 strips of 64 fused columns
  gemm_wmma<<<dim3(625, 16), 128, 0, stream>>>(xh, WhT, bcat, qb, kb, vb, out);

  esum_kernel  <<<dim3(EE*HH/256),    256, 0, stream>>>(edata, WeSum, es);
  logits_kernel<<<dim3(EE/8),         256, 0, stream>>>(qb, kb, es, esrc, edst, lg, me);
  exp_kernel   <<<dim3(EE*BB*HH/256), 256, 0, stream>>>(edst, me, lg, zb);
  agg_kernel   <<<dim3(EE/8),         256, 0, stream>>>(vb, lg, zb, esrc, edst, out);
}
